// differentiableTokenizerVisionTransformer_60473139527756
// MI455X (gfx1250) — matrix operand order repaired and checked by reading, not compiled, not run on hardware
//
#include <hip/hip_runtime.h>
#include <hip/hip_fp16.h>

typedef __attribute__((ext_vector_type(16))) _Float16 v16h;
typedef __attribute__((ext_vector_type(8)))  _Float16 v8h;
typedef __attribute__((ext_vector_type(8)))  float    v8f;

#define NBATCH 64
#define NCH    3
#define PIX    (224 * 224)        // 50176
#define SEG    196
#define MAXP   400
#define DIM    768
#define KDIM   1200               // C * MAXP
#define KPAD   1216               // padded to multiple of 32
#define CHUNK  1024
#define NCHUNK 49                 // 49 * 1024 == 50176
#define MROWS  (NBATCH * SEG)     // 12544

// ---------------- Pass 0: zero the f16 feats buffer ----------------
__global__ void k_zero_feats(int4* feats4) {
    int i = blockIdx.x * blockDim.x + threadIdx.x;   // one int4 (8 halves) each
    feats4[i] = make_int4(0, 0, 0, 0);
}

// ---------------- Pass 1: pack W^T into f16, K padded ----------------
__global__ void k_pack_wt(const float* __restrict__ W, _Float16* __restrict__ Wt) {
    int i = blockIdx.x * blockDim.x + threadIdx.x;   // over DIM * KPAD
    int d = i / KPAD;
    int k = i - d * KPAD;
    float v = (k < KDIM) ? W[(size_t)k * DIM + d] : 0.0f;
    Wt[i] = (_Float16)v;
}

// ---------------- Pass 2: per-chunk segment histograms ----------------
__global__ void k_hist(const int* __restrict__ seg, int* __restrict__ hist) {
    __shared__ int h[SEG];
    int b = blockIdx.x / NCHUNK;
    int c = blockIdx.x - b * NCHUNK;
    int tid = threadIdx.x;
    if (tid < SEG) h[tid] = 0;
    __syncthreads();
    const int* sp = seg + (size_t)b * PIX + c * CHUNK;
    #pragma unroll
    for (int i = 0; i < CHUNK / 256; ++i) {
        int s = sp[tid + i * 256];
        atomicAdd(&h[s], 1);
    }
    __syncthreads();
    if (tid < SEG) hist[(size_t)blockIdx.x * SEG + tid] = h[tid];
}

// ---------------- Pass 3: exclusive scan over chunks per segment ----------------
__global__ void k_scan(const int* __restrict__ hist, int* __restrict__ base) {
    int b = blockIdx.x;
    int s = threadIdx.x;
    if (s >= SEG) return;
    int run = 0;
    for (int c = 0; c < NCHUNK; ++c) {
        size_t idx = ((size_t)(b * NCHUNK + c)) * SEG + s;
        int t = hist[idx];
        base[idx] = run;
        run += t;
    }
}

// ---------------- Pass 4: ordered scatter of pixels into feats (f16) ----------------
// One wave (32 threads) per chunk; pixels processed in row-major order so ranks
// exactly match the reference's stable argsort.
__global__ void k_scatter(const float* __restrict__ img,
                          const int* __restrict__ seg,
                          const int* __restrict__ base,
                          _Float16* __restrict__ feats) {
    __shared__ int cnt[SEG];
    int b = blockIdx.x / NCHUNK;
    int c = blockIdx.x - b * NCHUNK;
    int lane = threadIdx.x;   // blockDim.x == 32
    for (int s = lane; s < SEG; s += 32)
        cnt[s] = base[(size_t)blockIdx.x * SEG + s];
    __syncthreads();

    const unsigned below = (1u << lane) - 1u;
    for (int g = 0; g < CHUNK / 32; ++g) {
        int p = c * CHUNK + g * 32 + lane;                 // pixel index in image
        int s = seg[(size_t)b * PIX + p];
        // lanes with identical segment id (8 bit-sliced ballots, wave32)
        unsigned mask = 0xffffffffu;
        #pragma unroll
        for (int k = 0; k < 8; ++k) {
            unsigned bit = (s >> k) & 1;
            unsigned bal = (unsigned)__ballot(bit);
            mask &= bit ? bal : ~bal;
        }
        int local  = __popc(mask & below);                 // in-order rank inside group
        int leader = __ffs(mask) - 1;
        int nmatch = __popc(mask);
        int base_r = 0;
        if (local == 0)
            base_r = atomicAdd(&cnt[s], nmatch);           // leaders have distinct s
        base_r = __shfl(base_r, leader, 32);
        int rank = base_r + local;
        if (rank < MAXP) {
            size_t row = ((size_t)b * SEG + s) * KPAD;
            #pragma unroll
            for (int ch = 0; ch < NCH; ++ch) {
                float v = img[((size_t)(b * NCH + ch)) * PIX + p];
                feats[row + ch * MAXP + rank] = (_Float16)v;
            }
        }
    }
}

// ---------------- Pass 5: WMMA GEMM  out[12544,768] = feats @ W + b ----------------
// Per wave: one 16(M) x 64(N) strip; A fragment reused across 4 N-tiles.
__global__ void k_gemm(const _Float16* __restrict__ feats,
                       const _Float16* __restrict__ Wt,
                       const float* __restrict__ bias,
                       float* __restrict__ out) {
    const int NT64 = DIM / 64;                 // 12
    int wave  = blockIdx.x * (blockDim.x >> 5) + (threadIdx.x >> 5);
    int mtile = wave / NT64;                   // 0..783
    int n64   = wave - mtile * NT64;           // 0..11
    int lane  = threadIdx.x & 31;
    int lm    = lane & 15;
    int lh    = lane >> 4;

    int arow  = mtile * 16 + lm;
    int nbase = n64 * 64;

    float bv[4];
    #pragma unroll
    for (int nt = 0; nt < 4; ++nt)
        bv[nt] = bias[nbase + nt * 16 + lm];

    v8f acc[4] = {};
    const _Float16* arowp = feats + (size_t)arow * KPAD;

    for (int kt = 0; kt < KPAD / 32; ++kt) {
        const _Float16* ap = arowp + kt * 32;
        union { v16h v; v8h h[2]; } ua;
        ua.h[0] = *(const v8h*)(ap + lh * 8);          // K = lh*8 .. +7
        ua.h[1] = *(const v8h*)(ap + 16 + lh * 8);     // K = 16+lh*8 .. +7
        __builtin_prefetch(ap + 32, 0, 1);             // next A tile -> global_prefetch
        #pragma unroll
        for (int nt = 0; nt < 4; ++nt) {
            const _Float16* bp = Wt + (size_t)(nbase + nt * 16 + lm) * KPAD
                                   + kt * 32 + lh * 8;
            union { v16h v; v8h h[2]; } ub;
            ub.h[0] = *(const v8h*)bp;                 // K = lh*8 .. +7
            ub.h[1] = *(const v8h*)(bp + 16);          // K = 16+lh*8 .. +7
            v16h bfrag = ub.v;
            acc[nt] = __builtin_amdgcn_wmma_f32_16x16x32_f16(
                false, ua.v, false, bfrag, (short)0, acc[nt], false, false);
        }
    }

    #pragma unroll
    for (int nt = 0; nt < 4; ++nt) {
        int col = nbase + nt * 16 + lm;
        #pragma unroll
        for (int g = 0; g < 8; ++g) {
            int row = mtile * 16 + g + 8 * lh;         // C/D layout: VGPR g, lane half
            out[(size_t)row * DIM + col] = acc[nt][g] + bv[nt];
        }
    }
}

extern "C" void kernel_launch(void* const* d_in, const int* in_sizes, int n_in,
                              void* d_out, int out_size, void* d_ws, size_t ws_size,
                              hipStream_t stream) {
    const float* img = (const float*)d_in[0];
    const int*   seg = (const int*)d_in[1];
    const float* W   = (const float*)d_in[2];
    const float* bia = (const float*)d_in[3];
    float* out = (float*)d_out;

    char* ws = (char*)d_ws;
    const size_t FEATS_BYTES = (size_t)MROWS * KPAD * sizeof(_Float16);   // 30,507,008
    const size_t WT_BYTES    = (size_t)DIM * KPAD * sizeof(_Float16);     //  1,867,776
    const size_t HIST_BYTES  = (size_t)NBATCH * NCHUNK * SEG * sizeof(int);
    _Float16* feats = (_Float16*)(ws);
    _Float16* Wt    = (_Float16*)(ws + FEATS_BYTES);
    int*      hist  = (int*)(ws + FEATS_BYTES + WT_BYTES);
    int*      base  = (int*)(ws + FEATS_BYTES + WT_BYTES + HIST_BYTES);

    // 0: zero feats (int4 = 8 halves per thread)
    int n16 = (int)(FEATS_BYTES / 16);
    k_zero_feats<<<n16 / 256, 256, 0, stream>>>((int4*)feats);
    // 1: pack W^T -> f16 (padded)
    k_pack_wt<<<(DIM * KPAD) / 256, 256, 0, stream>>>(W, Wt);
    // 2: chunk histograms
    k_hist<<<NBATCH * NCHUNK, 256, 0, stream>>>(seg, hist);
    // 3: per-segment exclusive scan over chunks
    k_scan<<<NBATCH, 256, 0, stream>>>(hist, base);
    // 4: ordered scatter into feats
    k_scatter<<<NBATCH * NCHUNK, 32, 0, stream>>>(img, seg, base, feats);
    // 5: WMMA GEMM + bias
    int waves = (MROWS / 16) * (DIM / 64);     // 784 * 12 = 9408
    k_gemm<<<waves / 8, 256, 0, stream>>>(feats, Wt, bia, out);
}
